// TopKRouter_72773925864231
// MI455X (gfx1250) — compile-verified, matrix operand order checked by reading, not run on hardware
//
#include <hip/hip_runtime.h>

// ---- problem constants (shapes fixed by the reference harness) ----
#define D      2048            // hidden dim (K)
#define E      64              // experts (N)
#define KC     32              // K-chunk per WMMA step
#define NCH    (D / KC)        // 64 chunks
#define MBLK   128             // tokens per workgroup (8 waves x 16 rows)
#define LDSTR  40              // padded bf16 stride per expert col (80B: 16B-aligned, 16 distinct banks)
#define BUFSZ  (E * LDSTR)     // bf16 elements per stage buffer (5120 B)

typedef __attribute__((ext_vector_type(16))) __bf16 v16bf;
typedef __attribute__((ext_vector_type(8)))  __bf16 bf16x8;
typedef __attribute__((ext_vector_type(4)))  __bf16 bf16x4;
typedef __attribute__((ext_vector_type(8)))  float  v8f;

// Split f32 into hi (RNE bf16) + lo (bf16 residual): 3-term bf16 WMMA ~= fp32 accuracy.
__device__ __forceinline__ void split_bf16(float f, __bf16& hi, __bf16& lo) {
    hi = (__bf16)f;
    lo = (__bf16)(f - (float)hi);
}

// Convert one float4 of w into hi/lo bf16x4 and store to LDS (8B ds stores).
__device__ __forceinline__ void stage4(const float4 v, __bf16* ph, __bf16* pl) {
    union { bf16x4 v4; __bf16 e[4]; } h, l;
    split_bf16(v.x, h.e[0], l.e[0]);
    split_bf16(v.y, h.e[1], l.e[1]);
    split_bf16(v.z, h.e[2], l.e[2]);
    split_bf16(v.w, h.e[3], l.e[3]);
    *(bf16x4*)ph = h.v4;
    *(bf16x4*)pl = l.v4;
}

__global__ __launch_bounds__(256)
void moe_logits_wmma(const float* __restrict__ x, const float* __restrict__ w,
                     float* __restrict__ out) {
    __shared__ __align__(16) __bf16 sBh[2 * BUFSZ];   // w chunk hi, ping-pong
    __shared__ __align__(16) __bf16 sBl[2 * BUFSZ];   // w chunk lo, ping-pong

    const int t    = threadIdx.x;
    const int wave = t >> 5;                 // wave32 (gfx1250)
    const int lane = t & 31;
    const int l16  = lane & 15;
    const int kb   = (lane >> 4) << 3;       // 0 or 8: K sub-offset per half-wave
    const size_t rowA = (size_t)blockIdx.x * MBLK + (size_t)wave * 16 + l16;
    const float* __restrict__ xrow = x + rowA * D;

    // staging ownership: thread covers float4 tiles t and t+256 of the 64x32 w chunk
    const int e0  = t >> 3;                  // expert rows 0..31
    const int kq0 = (t & 7) << 2;            // 0,4,...,28
    const int e1  = e0 + 32;                 // expert rows 32..63
    const float* __restrict__ wp0 = w + (size_t)e0 * D + kq0;
    const float* __restrict__ wp1 = w + (size_t)e1 * D + kq0;

    v8f acc0 = {}, acc1 = {}, acc2 = {}, acc3 = {};   // 16x16 f32 C tiles, N = 0..63

    // ---- prologue: stage chunk 0 into buffer 0; prefetch w chunk 1 and A chunk 0 ----
    float4 w0 = *(const float4*)(wp0);
    float4 w1 = *(const float4*)(wp1);
    stage4(w0, &sBh[e0 * LDSTR + kq0], &sBl[e0 * LDSTR + kq0]);
    stage4(w1, &sBh[e1 * LDSTR + kq0], &sBl[e1 * LDSTR + kq0]);
    w0 = *(const float4*)(wp0 + KC);
    w1 = *(const float4*)(wp1 + KC);
    float4 a0 = *(const float4*)(xrow + kb + 0);
    float4 a1 = *(const float4*)(xrow + kb + 4);
    float4 a2 = *(const float4*)(xrow + kb + 16);
    float4 a3 = *(const float4*)(xrow + kb + 20);
    __syncthreads();

    for (int i = 0; i < NCH; ++i) {
        const int curO = (i & 1) ? BUFSZ : 0;
        const int nxtO = (i & 1) ? 0 : BUFSZ;

        // ---- convert A regs (chunk i) into hi/lo fragments ----
        // 16-bit A striping: lanes 0-15 hold K = {0..7,16..23}, lanes 16-31 K = {8..15,24..31}
        union { float4 q[4]; float f[16]; } af;
        af.q[0] = a0; af.q[1] = a1; af.q[2] = a2; af.q[3] = a3;
        union { v16bf v; __bf16 e[16]; } ah, al;
        #pragma unroll
        for (int j = 0; j < 16; ++j) split_bf16(af.f[j], ah.e[j], al.e[j]);

        // ---- prefetch A for chunk i+1 (hidden under WMMA + barrier) ----
        if (i + 1 < NCH) {
            const float* ap = xrow + (i + 1) * KC + kb;
            a0 = *(const float4*)(ap + 0);
            a1 = *(const float4*)(ap + 4);
            a2 = *(const float4*)(ap + 16);
            a3 = *(const float4*)(ap + 20);
        }

        // ---- stage chunk i+1 into the other buffer; prefetch w chunk i+2 ----
        if (i + 1 < NCH) {
            stage4(w0, &sBh[nxtO + e0 * LDSTR + kq0], &sBl[nxtO + e0 * LDSTR + kq0]);
            stage4(w1, &sBh[nxtO + e1 * LDSTR + kq0], &sBl[nxtO + e1 * LDSTR + kq0]);
            if (i + 2 < NCH) {
                w0 = *(const float4*)(wp0 + (size_t)(i + 2) * KC);
                w1 = *(const float4*)(wp1 + (size_t)(i + 2) * KC);
            }
        }

        // ---- 4 N-tiles x 3 split-term WMMAs from buffer `cur` ----
        #pragma unroll
        for (int nt = 0; nt < 4; ++nt) {
            const int base = curO + (nt * 16 + l16) * LDSTR + kb;
            union { v16bf v; bf16x8 h[2]; } bh, bl;
            bh.h[0] = *(const bf16x8*)(&sBh[base]);
            bh.h[1] = *(const bf16x8*)(&sBh[base + 16]);
            bl.h[0] = *(const bf16x8*)(&sBl[base]);
            bl.h[1] = *(const bf16x8*)(&sBl[base + 16]);
            v8f* acc = (nt == 0) ? &acc0 : (nt == 1) ? &acc1 : (nt == 2) ? &acc2 : &acc3;
            *acc = __builtin_amdgcn_wmma_f32_16x16x32_bf16(false, ah.v, false, bh.v,
                                                           (short)0, *acc, false, false);
            *acc = __builtin_amdgcn_wmma_f32_16x16x32_bf16(false, ah.v, false, bl.v,
                                                           (short)0, *acc, false, false);
            *acc = __builtin_amdgcn_wmma_f32_16x16x32_bf16(false, al.v, false, bh.v,
                                                           (short)0, *acc, false, false);
        }
        __syncthreads();   // one barrier per chunk: protects ping-pong overwrite + visibility
    }

    // ---- store logits: C layout: VGPR v, lane l -> row = v + 8*(l>=16), col = l&15 ----
    const size_t rowC0 = (size_t)blockIdx.x * MBLK + (size_t)wave * 16 + 8 * (lane >> 4);
    #pragma unroll
    for (int v = 0; v < 8; ++v) {
        float* o = out + (rowC0 + v) * E + l16;
        o[0]  = acc0[v];
        o[16] = acc1[v];
        o[32] = acc2[v];
        o[48] = acc3[v];
    }
}

__global__ __launch_bounds__(256)
void softmax_top2(float* __restrict__ out, int T) {
    const int tk = blockIdx.x * 256 + threadIdx.x;
    if (tk >= T) return;
    float* row = out + (size_t)tk * E;   // logits in place, L2-hot

    float l[E];
    float m = -3.0e38f;
    #pragma unroll
    for (int i = 0; i < E; i += 4) {
        float4 v = *(const float4*)(row + i);
        l[i] = v.x; l[i+1] = v.y; l[i+2] = v.z; l[i+3] = v.w;
        m = fmaxf(m, fmaxf(fmaxf(v.x, v.y), fmaxf(v.z, v.w)));
    }
    float s = 0.0f;
    #pragma unroll
    for (int i = 0; i < E; ++i) { l[i] = expf(l[i] - m); s += l[i]; }
    const float inv = 1.0f / s;

    float v1 = -1.0f, v2 = -1.0f; int i1 = 0, i2 = 0;
    #pragma unroll
    for (int i = 0; i < E; ++i) {
        const float p = l[i] * inv;
        l[i] = p;
        if (p > v1)      { v2 = v1; i2 = i1; v1 = p; i1 = i; }  // strict >: first index wins ties
        else if (p > v2) { v2 = p;  i2 = i; }
    }
    #pragma unroll
    for (int i = 0; i < E; i += 4) {
        float4 v = { l[i], l[i+1], l[i+2], l[i+3] };
        *(float4*)(row + i) = v;
    }
    float* tv = out + (size_t)T * E;            // topk_val region
    float* ti = tv + (size_t)T * 2;             // topk_idx region (float-valued indices)
    tv[2*tk]     = v1;  tv[2*tk + 1] = v2;
    ti[2*tk]     = (float)i1;
    ti[2*tk + 1] = (float)i2;
}

extern "C" void kernel_launch(void* const* d_in, const int* in_sizes, int n_in,
                              void* d_out, int out_size, void* d_ws, size_t ws_size,
                              hipStream_t stream) {
    const float* x = (const float*)d_in[0];   // [4,8192,2048] f32
    const float* w = (const float*)d_in[1];   // [64,2048]     f32
    float* out = (float*)d_out;               // probs | topk_val | topk_idx(float)
    const int T = in_sizes[0] / D;            // 32768 tokens

    moe_logits_wmma<<<T / MBLK, 256, 0, stream>>>(x, w, out);
    softmax_top2<<<(T + 255) / 256, 256, 0, stream>>>(out, T);
}
